// DetectionLoss_87617332839044
// MI455X (gfx1250) — compile-verified, matrix operand order checked by reading, not run on hardware
//
#include <hip/hip_runtime.h>

typedef __attribute__((ext_vector_type(16))) _Float16 v16h;
typedef __attribute__((ext_vector_type(8)))  float    v8f;

#define OMAX 64

__device__ __forceinline__ float smooth_l1(float d) {
    float ad = fabsf(d);
    return ad < 1.0f ? 0.5f * ad * ad : ad - 0.5f;
}

// monotonic float -> uint key (order-preserving for all floats incl. negatives)
__device__ __forceinline__ unsigned f2key(float v) {
    unsigned b = __float_as_uint(v);
    return (b & 0x80000000u) ? ~b : (b | 0x80000000u);
}
__device__ __forceinline__ float key2f(unsigned k) {
    unsigned b = (k & 0x80000000u) ? (k & 0x7FFFFFFFu) : ~k;
    return __uint_as_float(b);
}

__global__ void dl_init(unsigned* su, float* sf) {
    su[0] = 0u; su[1] = 0u;           // n_pos, n_neg
    sf[0] = 0.f; sf[1] = 0.f; sf[2] = 0.f;  // box_sum, sum_pos, wsum_pos
}

// One thread per (b, a): matching, box loss, CE, neg-mining key.
__global__ void __launch_bounds__(256)
dl_match(const float* __restrict__ pred_boxes,
         const float* __restrict__ pred_classes,
         const float* __restrict__ true_boxes,
         const int*   __restrict__ true_classes,
         const float* __restrict__ anchors,
         unsigned* __restrict__ scal_u,
         float*    __restrict__ scal_f,
         unsigned* __restrict__ neg_keys,
         int A, int O)
{
    __shared__ float    tb[OMAX * 4];
    __shared__ int      tc[OMAX];
    __shared__ _Float16 red[3][512];

    const int t = threadIdx.x;
    const int b = blockIdx.y;
    const int a = blockIdx.x * blockDim.x + t;
    const bool active = (a < A);

    if (t < O * 4) tb[t] = true_boxes[b * O * 4 + t];
    if (t < O)     tc[t] = true_classes[b * O + t];
    __syncthreads();

    unsigned npos_l = 0, nneg_l = 0;
    float box_l = 0.f, spos_l = 0.f, wpos_l = 0.f;

    if (active) {
        __builtin_prefetch(&pred_classes[(size_t)(b * A + a) * 2], 0, 3);

        // anchor corner form
        const float acx = anchors[a * 4 + 0], acy = anchors[a * 4 + 1];
        const float aw  = anchors[a * 4 + 2], ah  = anchors[a * 4 + 3];
        const float ax1 = acx - aw * 0.5f, ay1 = acy - ah * 0.5f;
        const float ax2 = acx + aw * 0.5f, ay2 = acy + ah * 0.5f;
        const float area_a = (ax2 - ax1) * (ay2 - ay1);

        // pass 1: best IoU over gt objects (pad -> -1)
        float best = -INFINITY;
        for (int o = 0; o < O; ++o) {
            float ov = -1.0f;
            if (tc[o] >= 0) {
                const float x1 = tb[o*4+0], y1 = tb[o*4+1], x2 = tb[o*4+2], y2 = tb[o*4+3];
                const float iw = fminf(ax2, x2) - fmaxf(ax1, x1);
                const float ih = fminf(ay2, y2) - fmaxf(ay1, y1);
                const float inter = fmaxf(iw, 0.f) * fmaxf(ih, 0.f);
                const float area_b = (x2 - x1) * (y2 - y1);
                ov = inter / (area_a + area_b - inter);
            }
            best = fmaxf(best, ov);
        }
        const bool negative = best < 0.5f;

        // 2-class log-softmax
        const float p0 = pred_classes[(size_t)(b * A + a) * 2 + 0];
        const float p1 = pred_classes[(size_t)(b * A + a) * 2 + 1];
        const float mx = fmaxf(p0, p1);
        const float lse = mx + logf(expf(p0 - mx) + expf(p1 - mx));
        const float logp0 = p0 - lse, logp1 = p1 - lse;

        const size_t pb = (size_t)(b * A + a) * 4;
        const float pbx0 = pred_boxes[pb + 0], pbx1 = pred_boxes[pb + 1];
        const float pbx2 = pred_boxes[pb + 2], pbx3 = pred_boxes[pb + 3];

        // pass 2: positives (recompute IoU; cheaper than 32 live regs)
        for (int o = 0; o < O; ++o) {
            float ov = -1.0f;
            if (tc[o] >= 0) {
                const float x1 = tb[o*4+0], y1 = tb[o*4+1], x2 = tb[o*4+2], y2 = tb[o*4+3];
                const float iw = fminf(ax2, x2) - fmaxf(ax1, x1);
                const float ih = fminf(ay2, y2) - fmaxf(ay1, y1);
                const float inter = fmaxf(iw, 0.f) * fmaxf(ih, 0.f);
                const float area_b = (x2 - x1) * (y2 - y1);
                ov = inter / (area_a + area_b - inter);
            }
            if (fabsf(best - ov) < 1e-6f && ov > 0.5f) {
                ++npos_l;
                const float x1 = tb[o*4+0], y1 = tb[o*4+1], x2 = tb[o*4+2], y2 = tb[o*4+3];
                const float gcx = ((x1 + x2) * 0.5f - acx) / (0.1f * aw);
                const float gcy = ((y1 + y2) * 0.5f - acy) / (0.1f * ah);
                const float gw  = logf((x2 - x1) / aw) / 0.2f;
                const float gh  = logf((y2 - y1) / ah) / 0.2f;
                box_l += smooth_l1(pbx0 - gcx) + smooth_l1(pbx1 - gcy)
                       + smooth_l1(pbx2 - gw)  + smooth_l1(pbx3 - gh);
                int cls = tc[o]; cls = cls < 0 ? 0 : (cls > 1 ? 1 : cls);
                const float w  = (cls == 1) ? 4.0f : 1.0f;
                const float ce = (cls == 1) ? -logp1 : -logp0;
                spos_l += w * ce;
                wpos_l += w;
            }
        }

        nneg_l = negative ? 1u : 0u;
        // sentinel (-1.0) sorts below every valid CE (>= 0); k <= n_neg so never taken
        neg_keys[(size_t)b * A + a] = f2key(negative ? -logp0 : -1.0f);
    }

    // ---- integer counts: wave32 butterfly + one atomic per wave ----
    unsigned np = npos_l, nn = nneg_l;
    #pragma unroll
    for (int off = 16; off > 0; off >>= 1) {
        np += __shfl_xor(np, off, 32);
        nn += __shfl_xor(nn, off, 32);
    }
    if ((t & 31) == 0) {
        atomicAdd(&scal_u[0], np);
        atomicAdd(&scal_u[1], nn);
    }

    // ---- float block sums through the matrix pipe: D = A_tile x ones ----
    // each f32 partial split into (hi,lo) f16 pair -> 512 f16 summands per tile
    float part[3] = { box_l, spos_l, wpos_l };
    #pragma unroll
    for (int q = 0; q < 3; ++q) {
        const float x = part[q];
        const _Float16 hi = (_Float16)x;
        const _Float16 lo = (_Float16)(x - (float)hi);
        red[q][t]       = hi;
        red[q][256 + t] = lo;
    }
    __syncthreads();

    if (t < 32) {  // wave 0, all 32 lanes active -> EXEC all ones for WMMA
        v16h ones;
        #pragma unroll
        for (int i = 0; i < 16; ++i) ones[i] = (_Float16)1.0f;
        #pragma unroll
        for (int q = 0; q < 3; ++q) {
            v16h am;
            #pragma unroll
            for (int i = 0; i < 16; ++i) am[i] = red[q][t * 16 + i];
            v8f c = {};
            // with B = all-ones: D[m][n] = rowsum_m(A), independent of A lane layout
            c = __builtin_amdgcn_wmma_f32_16x16x32_f16(
                    false, am, false, ones, (short)0, c, false, false);
            float s = c[0] + c[1] + c[2] + c[3] + c[4] + c[5] + c[6] + c[7];
            // column 0 lives in lane 0 (rows 0-7) and lane 16 (rows 8-15)
            float tot = __shfl(s, 0, 32) + __shfl(s, 16, 32);
            if (t == 0) atomicAdd(&scal_f[q], tot);
        }
    }
}

// Single workgroup: exact top-k sum via 4-pass 8-bit radix select, then finalize.
__global__ void __launch_bounds__(1024)
dl_select_finalize(const unsigned* __restrict__ neg_keys,
                   const unsigned* __restrict__ scal_u,
                   const float*    __restrict__ scal_f,
                   float* __restrict__ out, int N)
{
    __shared__ unsigned hist[256];
    __shared__ unsigned sh_prefix, sh_krem;
    __shared__ float s_red[1024];

    const int t = threadIdx.x;
    const unsigned n_pos = scal_u[0];
    const unsigned n_neg = scal_u[1];
    unsigned long long k = 10ull * (unsigned long long)n_pos;
    if (k > n_neg) k = n_neg;

    float sum_neg = 0.0f;
    if (k > 0) {
        if (t == 0) { sh_prefix = 0u; sh_krem = (unsigned)k; }
        __syncthreads();
        for (int shift = 24; shift >= 0; shift -= 8) {
            if (t < 256) hist[t] = 0u;
            __syncthreads();
            const unsigned prefix = sh_prefix;
            for (int i = t; i < N; i += 1024) {
                const unsigned key = neg_keys[i];
                if (((unsigned long long)(key ^ prefix) >> (shift + 8)) == 0ull)
                    atomicAdd(&hist[(key >> shift) & 0xFFu], 1u);
            }
            __syncthreads();
            if (t == 0) {
                unsigned krem = sh_krem, running = 0u, chosen = 0u;
                for (int bin = 255; bin >= 0; --bin) {
                    const unsigned c = hist[bin];
                    if (running + c >= krem) { chosen = (unsigned)bin; sh_krem = krem - running; break; }
                    running += c;
                }
                sh_prefix = prefix | (chosen << shift);
            }
            __syncthreads();
        }
        const unsigned tkey = sh_prefix;   // exact k-th largest key
        const unsigned krem = sh_krem;     // how many copies of tkey to take
        float local = 0.0f;
        for (int i = t; i < N; i += 1024) {
            const unsigned key = neg_keys[i];
            if (key > tkey) local += key2f(key);
        }
        s_red[t] = local;
        __syncthreads();
        for (int off = 512; off > 0; off >>= 1) {
            if (t < off) s_red[t] += s_red[t + off];
            __syncthreads();
        }
        sum_neg = s_red[0] + (float)krem * key2f(tkey);
    }

    if (t == 0) {
        const float denom = (float)(n_pos > 0u ? n_pos : 1u);
        const float box_loss = scal_f[0] / denom;
        const float kf = (float)k;
        const float cls_loss = 10.0f * (scal_f[1] + sum_neg)
                             / fmaxf(scal_f[2] + kf, 1e-6f) / denom;
        out[0] = box_loss;
        out[1] = cls_loss;
        out[2] = box_loss + cls_loss;
    }
}

extern "C" void kernel_launch(void* const* d_in, const int* in_sizes, int n_in,
                              void* d_out, int out_size, void* d_ws, size_t ws_size,
                              hipStream_t stream) {
    const float* pred_boxes   = (const float*)d_in[0];
    const float* pred_classes = (const float*)d_in[1];
    const float* true_boxes   = (const float*)d_in[2];
    const int*   true_classes = (const int*)d_in[3];
    const float* anchors      = (const float*)d_in[4];

    const int A = in_sizes[4] / 4;            // 16384
    const int B = in_sizes[0] / (A * 4);      // 8
    const int O = in_sizes[3] / B;            // 32
    const int N = B * A;

    unsigned* scal_u   = (unsigned*)d_ws;                      // 2 u32
    float*    scal_f   = (float*)((char*)d_ws + 8);            // 3 f32
    unsigned* neg_keys = (unsigned*)((char*)d_ws + 64);        // N u32

    dl_init<<<1, 1, 0, stream>>>(scal_u, scal_f);

    dim3 grid((A + 255) / 256, B);
    dl_match<<<grid, 256, 0, stream>>>(pred_boxes, pred_classes, true_boxes,
                                       true_classes, anchors,
                                       scal_u, scal_f, neg_keys, A, O);

    dl_select_finalize<<<1, 1024, 0, stream>>>(neg_keys, scal_u, scal_f,
                                               (float*)d_out, N);
}
